// SelfAttention_46368466927627
// MI455X (gfx1250) — matrix operand order repaired and checked by reading, not compiled, not run on hardware
//
#include <hip/hip_runtime.h>
#include <hip/hip_bf16.h>

// ---------------------------------------------------------------------------
// Self-attention (B=32, N=1024, C=256, d=32) for MI455X / gfx1250.
// All matrix math through v_wmma_f32_16x16x32_bf16 (f32 accumulate).
// Flash-style online softmax; K/V^T chunks staged to LDS with double-buffered
// GLOBAL_LOAD_ASYNC_TO_LDS (ASYNCcnt) when the builtins are available.
// ---------------------------------------------------------------------------

typedef __attribute__((ext_vector_type(16))) __bf16 bf16x16;
typedef __attribute__((ext_vector_type(8)))  __bf16 bf16x8;
typedef __attribute__((ext_vector_type(8)))  float  f32x8;
typedef int v2i __attribute__((vector_size(8)));
typedef int v4i __attribute__((vector_size(16)));

#define WMMA_BF16(A, B, C) \
  __builtin_amdgcn_wmma_f32_16x16x32_bf16(false, (A), false, (B), (short)0, (C), false, false)

#if defined(__gfx1250__) && \
    __has_builtin(__builtin_amdgcn_global_load_async_to_lds_b64) && \
    __has_builtin(__builtin_amdgcn_global_load_async_to_lds_b128) && \
    __has_builtin(__builtin_amdgcn_s_wait_asynccnt)
#define USE_ASYNC 1
#else
#define USE_ASYNC 0
#endif

#define GLOBAL_AS __attribute__((address_space(1)))
#define LDS_AS    __attribute__((address_space(3)))

static __device__ __forceinline__ bf16x8 ld8(const __bf16* p) {
  return *(const bf16x8*)p;
}
static __device__ __forceinline__ bf16x16 cat8(bf16x8 lo, bf16x8 hi) {
  return __builtin_shufflevector(lo, hi, 0,1,2,3,4,5,6,7,8,9,10,11,12,13,14,15);
}
// A-matrix 16x32 bf16 fragment: lane row = lane%16 (caller bakes into rowPtr),
// K chunks at {0..7,16..23} (lanes 0-15) / {8..15,24..31} (lanes 16-31).
static __device__ __forceinline__ bf16x16 loadA(const __bf16* rowPtr, int lane) {
  const int k0 = (lane < 16) ? 0 : 8;
  return cat8(ld8(rowPtr + k0), ld8(rowPtr + k0 + 16));
}
// B-matrix 32x16 bf16 fragment from B^T rows: lane col = lane%16 (in colPtr),
// K chunks at {0..7,16..23} (lanes 0-15) / {8..15,24..31} (lanes 16-31).
static __device__ __forceinline__ bf16x16 loadB(const __bf16* colPtr, int lane) {
  const int k0 = (lane < 16) ? 0 : 8;
  return cat8(ld8(colPtr + k0), ld8(colPtr + k0 + 16));
}

// ---------------------------------------------------------------------------
// Prep kernels.
// ---------------------------------------------------------------------------
__global__ void __launch_bounds__(256)
prep_weights(const float* __restrict__ Wq, const float* __restrict__ bq,
             const float* __restrict__ Wk, const float* __restrict__ bk,
             const float* __restrict__ Wv, const float* __restrict__ bv,
             __bf16* __restrict__ WcatT, float* __restrict__ bcat) {
  const int idx = blockIdx.x * 256 + threadIdx.x;
  if (idx < 320 * 256) {
    const int j = idx >> 8, c = idx & 255;
    float w;
    if (j < 32)      w = Wq[c * 32 + j];
    else if (j < 64) w = Wk[c * 32 + (j - 32)];
    else             w = Wv[c * 256 + (j - 64)];
    WcatT[j * 256 + c] = (__bf16)w;
  }
  if (idx < 320) {
    bcat[idx] = (idx < 32) ? bq[idx] : (idx < 64) ? bk[idx - 32] : bv[idx - 64];
  }
}

__global__ void __launch_bounds__(256)
cvt_x(const float* __restrict__ x, __bf16* __restrict__ xb) {
  const int i = (blockIdx.x * 256 + threadIdx.x) * 4;
  const float4 f = *(const float4*)(x + i);
  __align__(8) __bf16 o[4] = {(__bf16)f.x, (__bf16)f.y, (__bf16)f.z, (__bf16)f.w};
  *(uint2*)(xb + i) = *(const uint2*)o;
}

// ---------------------------------------------------------------------------
// Projection GEMM: [32768 x 256] @ [256 x 320] -> qb(32, pre-scaled 1/sqrt(d)),
// kb(32) row-major, and V written TRANSPOSED: vT[b][ch][n] so the attention
// kernel can stage V^T chunks as contiguous async copies.
// ---------------------------------------------------------------------------
__global__ void __launch_bounds__(128)
proj_qkv(const __bf16* __restrict__ xb, const __bf16* __restrict__ WcatT,
         const float* __restrict__ bcat,
         __bf16* __restrict__ qb, __bf16* __restrict__ kbuf,
         __bf16* __restrict__ vT) {
  const int lane    = threadIdx.x & 31;
  const int wave    = threadIdx.x >> 5;
  const int rowTile = blockIdx.x;             // 0..2047
  const int colTile = blockIdx.y * 4 + wave;  // 0..19
  const int rowA = rowTile * 16 + (lane & 15);
  const int col  = colTile * 16 + (lane & 15);

  f32x8 acc = {};
  const __bf16* aRow = xb + (size_t)rowA * 256;
  const __bf16* bCol = WcatT + (size_t)col * 256;
#pragma unroll
  for (int kc = 0; kc < 8; ++kc) {
    bf16x16 a = loadA(aRow + kc * 32, lane);
    bf16x16 b = loadB(bCol + kc * 32, lane);
    acc = WMMA_BF16(a, b, acc);
  }
  const float bias  = bcat[col];
  const int   rbase = rowTile * 16 + ((lane >> 4) << 3);  // D-layout rows
#pragma unroll
  for (int v = 0; v < 8; ++v) {
    const int row = rbase + v;
    const float val = acc[v] + bias;
    if (col < 32) {
      qb[(size_t)row * 32 + col] = (__bf16)(val * 0.17677669529663687f);
    } else if (col < 64) {
      kbuf[(size_t)row * 32 + (col - 32)] = (__bf16)val;
    } else {
      const int bb = row >> 10, n = row & 1023, ch = col - 64;
      vT[((size_t)bb * 256 + ch) * 1024 + n] = (__bf16)val;  // transposed store
    }
  }
}

// ---------------------------------------------------------------------------
// Chunk staging: K chunk (32 keys x 32 d, 2 KB contiguous) and V^T chunk
// (256 ch x 32 keys, 64 B per ch row). Async-to-LDS when available.
// ---------------------------------------------------------------------------
static __device__ __forceinline__ void stage_chunk(
    const __bf16* __restrict__ kbuf, const __bf16* __restrict__ vT,
    __bf16* kDst, __bf16* vDst, int b, int kb0, int t) {
  const __bf16* kSrc = kbuf + (size_t)(b * 1024 + kb0) * 32;
  const __bf16* vSrc = vT + ((size_t)b * 256 + t) * 1024 + kb0;
#if USE_ASYNC
  __builtin_amdgcn_global_load_async_to_lds_b64(
      (GLOBAL_AS v2i*)(kSrc + t * 4), (LDS_AS v2i*)(kDst + t * 4), 0, 0);
#pragma unroll
  for (int i = 0; i < 4; ++i)
    __builtin_amdgcn_global_load_async_to_lds_b128(
        (GLOBAL_AS v4i*)(vSrc + i * 8), (LDS_AS v4i*)(vDst + t * 32 + i * 8),
        0, 0);
#else
  *(uint2*)(kDst + t * 4) = *(const uint2*)(kSrc + t * 4);
#pragma unroll
  for (int i = 0; i < 4; ++i)
    *(uint4*)(vDst + t * 32 + i * 8) = *(const uint4*)(vSrc + i * 8);
#endif
}

static __device__ __forceinline__ void wait_stage() {
#if USE_ASYNC
  __builtin_amdgcn_s_wait_asynccnt(0);
#endif
}

// ---------------------------------------------------------------------------
// Flash attention: 256 blocks x 256 threads (8 waves). Wave owns 16 query
// rows x all 256 channels (16 f32x8 accumulators). Keys swept in chunks of 32
// (= WMMA K), double-buffered LDS staging overlapped with compute.
// ---------------------------------------------------------------------------
__global__ void __launch_bounds__(256)
flash_attn(const __bf16* __restrict__ qb, const __bf16* __restrict__ kbuf,
           const __bf16* __restrict__ vT, const float* __restrict__ x,
           const float* __restrict__ gamma, float* __restrict__ out) {
  __shared__ __bf16 kLds[2][32 * 32];    // [key][d]  = B^T for q@k^T
  __shared__ __bf16 vLds[2][256 * 32];   // [ch][key] = B^T for P@V
  __shared__ __bf16 pLds[8][16 * 32];    // wave-private P staging

  const int t = threadIdx.x;
  const int lane = t & 31, wave = t >> 5;
  const int rowBase = blockIdx.x * 128 + wave * 16;  // global query row
  const int b = blockIdx.x >> 3;                     // batch (1024 rows / 128)
  const float g = gamma[0];

  const bf16x16 qfrag = loadA(qb + (size_t)(rowBase + (lane & 15)) * 32, lane);

  f32x8 O[16];
#pragma unroll
  for (int ct = 0; ct < 16; ++ct) O[ct] = (f32x8){};
  float m[8], l[8];
#pragma unroll
  for (int v = 0; v < 8; ++v) { m[v] = -3.0e38f; l[v] = 0.f; }

  // Prologue: stage chunk 0 into buffer 0.
  stage_chunk(kbuf, vT, kLds[0], vLds[0], b, 0, t);

  for (int kc = 0; kc < 32; ++kc) {
    const int cur = kc & 1;
    wait_stage();      // own wave's async copies for chunk kc complete
    __syncthreads();   // all waves' copies visible; prev buffer fully consumed
    if (kc + 1 < 32)   // prefetch next chunk into the other buffer
      stage_chunk(kbuf, vT, kLds[cur ^ 1], vLds[cur ^ 1], b, (kc + 1) * 32, t);

    // --- S = q @ k^T, two 16-key subtiles ---
    const bf16x16 kf0 = loadB(kLds[cur] + (size_t)(lane & 15) * 32, lane);
    const bf16x16 kf1 = loadB(kLds[cur] + (size_t)(16 + (lane & 15)) * 32, lane);
    f32x8 z = {};
    f32x8 s0 = WMMA_BF16(qfrag, kf0, z);
    f32x8 s1 = WMMA_BF16(qfrag, kf1, z);

    // --- online softmax: per-row (xor within 16-lane column groups) ---
    float corr[8];
#pragma unroll
    for (int v = 0; v < 8; ++v) {
      float tm = fmaxf(s0[v], s1[v]);
      tm = fmaxf(tm, __shfl_xor(tm, 1));
      tm = fmaxf(tm, __shfl_xor(tm, 2));
      tm = fmaxf(tm, __shfl_xor(tm, 4));
      tm = fmaxf(tm, __shfl_xor(tm, 8));
      const float mn = fmaxf(m[v], tm);
      corr[v] = __expf(m[v] - mn);
      const float p0 = __expf(s0[v] - mn);
      const float p1 = __expf(s1[v] - mn);
      s0[v] = p0; s1[v] = p1;
      float rs = p0 + p1;
      rs += __shfl_xor(rs, 1);
      rs += __shfl_xor(rs, 2);
      rs += __shfl_xor(rs, 4);
      rs += __shfl_xor(rs, 8);
      l[v] = l[v] * corr[v] + rs;
      m[v] = mn;
    }
#pragma unroll
    for (int ct = 0; ct < 16; ++ct)
#pragma unroll
      for (int v = 0; v < 8; ++v) O[ct][v] *= corr[v];

    // --- P: D-layout scatter to wave-private LDS, gather as A fragment ---
    {
      __bf16* p = &pLds[wave][0];
      const int colp = lane & 15, rb2 = (lane >> 4) << 3;
#pragma unroll
      for (int v = 0; v < 8; ++v) {
        p[(rb2 + v) * 32 + colp]      = (__bf16)s0[v];
        p[(rb2 + v) * 32 + 16 + colp] = (__bf16)s1[v];
      }
    }
    __syncthreads();  // orders pLds store->load (wave-private; barrier is uniform)
    const bf16x16 pfrag = loadA(&pLds[wave][(lane & 15) * 32], lane);

    // --- O += P @ V over all 16 channel tiles ---
#pragma unroll
    for (int ct = 0; ct < 16; ++ct) {
      const bf16x16 vf = loadB(vLds[cur] + (size_t)(ct * 16 + (lane & 15)) * 32, lane);
      O[ct] = WMMA_BF16(pfrag, vf, O[ct]);
    }
  }

  // --- epilogue: out = gamma * (O / l) + x ---
  const int colc = lane & 15;
  const int rb3  = rowBase + ((lane >> 4) << 3);
#pragma unroll
  for (int ct = 0; ct < 16; ++ct) {
#pragma unroll
    for (int v = 0; v < 8; ++v) {
      const size_t off = (size_t)(rb3 + v) * 256 + ct * 16 + colc;
      out[off] = g * (O[ct][v] / l[v]) + x[off];
    }
  }
}

// ---------------------------------------------------------------------------
// Workspace layout (bytes):
//   xb    @ 0         : 32768*256*2 = 16,777,216
//   vT    @ 16777216  : 16,777,216   (transposed: [b][ch][n])
//   qb    @ 33554432  :  2,097,152
//   kb    @ 35651584  :  2,097,152
//   WcatT @ 37748736  :    163,840
//   bcat  @ 37912576  :      1,280   (total ~37.9 MB)
// ---------------------------------------------------------------------------
extern "C" void kernel_launch(void* const* d_in, const int* in_sizes, int n_in,
                              void* d_out, int out_size, void* d_ws, size_t ws_size,
                              hipStream_t stream) {
  const float* x     = (const float*)d_in[0];
  const float* Wq    = (const float*)d_in[1];
  const float* bq    = (const float*)d_in[2];
  const float* Wk    = (const float*)d_in[3];
  const float* bk    = (const float*)d_in[4];
  const float* Wv    = (const float*)d_in[5];
  const float* bv    = (const float*)d_in[6];
  const float* gamma = (const float*)d_in[7];
  float* out = (float*)d_out;

  char* ws = (char*)d_ws;
  __bf16* xb    = (__bf16*)(ws);
  __bf16* vT    = (__bf16*)(ws + 16777216);
  __bf16* qb    = (__bf16*)(ws + 33554432);
  __bf16* kb    = (__bf16*)(ws + 35651584);
  __bf16* WcatT = (__bf16*)(ws + 37748736);
  float*  bcat  = (float*) (ws + 37912576);

  prep_weights<<<320, 256, 0, stream>>>(Wq, bq, Wk, bk, Wv, bv, WcatT, bcat);
  cvt_x<<<8192, 256, 0, stream>>>(x, xb);
  dim3 pgrid(2048, 5);
  proj_qkv<<<pgrid, 128, 0, stream>>>(xb, WcatT, bcat, qb, kb, vT);
  flash_attn<<<256, 256, 0, stream>>>(qb, kb, vT, x, gamma, out);
}
